// GenotypeHeteroEncoder_46995532152982
// MI455X (gfx1250) — compile-verified, matrix operand order checked by reading, not run on hardware
//
#include <hip/hip_runtime.h>
#include <hip/hip_bf16.h>

// GATv2 hetero-GNN forward for MI455X (gfx1250, wave32).
// Dense linears use v_wmma_f32_16x16x32_bf16 (bf16 A/B, f32 accum); all edge
// softmax/aggregation math is f32 with native f32 global atomics.
// Dead-output pruning: layer-2 only needs hap__gene; layer-1 only needs
// snp__hap, gene__hap, hap__gene (hap__snp never feeds the output).

typedef __bf16 bf16;
typedef __attribute__((ext_vector_type(16))) __bf16 v16bf;
typedef __attribute__((ext_vector_type(8)))  __bf16 v8bf;
typedef __attribute__((ext_vector_type(8)))  float  v8f;

#define HIDC 128
#define HC   256
#define ENC_NEG_INF 0x007FFFFFu

// ---------------- GEMM: C[N,M] = A[N,K](bf16) @ Bt[M,K]^T(bf16) + bias[M] ----
// One wave per 16x16 tile; block = (32,4) waves covering 4 adjacent col tiles.
// A-frag (16x32 bf16, lane m=lane%16): lanes<16 hold K k0+[0..7],k0+[16..23];
// lanes>=16 hold k0+[8..15],k0+[24..31]. Bt row-major gives the symmetric
// B-frag with two contiguous 16B loads per lane.
__global__ void gemm_bf16(const bf16* __restrict__ A,
                          const bf16* __restrict__ Bt,
                          const float* __restrict__ bias,
                          float* __restrict__ C,
                          int K, int M) {
  const int row0 = blockIdx.x * 16;
  const int col0 = (blockIdx.y * 4 + threadIdx.y) * 16;
  const int lane = threadIdx.x;
  const int lm = lane & 15;
  const int hh = (lane >> 4) & 1;
  const bf16* arow = A  + (size_t)(row0 + lm) * K + hh * 8;
  const bf16* brow = Bt + (size_t)(col0 + lm) * K + hh * 8;
  v8f acc = {};
  for (int k0 = 0; k0 < K; k0 += 32) {
    v8bf alo = *(const v8bf*)(arow + k0);
    v8bf ahi = *(const v8bf*)(arow + k0 + 16);
    v8bf blo = *(const v8bf*)(brow + k0);
    v8bf bhi = *(const v8bf*)(brow + k0 + 16);
    v16bf av, bv;
#pragma unroll
    for (int i = 0; i < 8; ++i) {
      av[i] = alo[i]; av[8 + i] = ahi[i];
      bv[i] = blo[i]; bv[8 + i] = bhi[i];
    }
    acc = __builtin_amdgcn_wmma_f32_16x16x32_bf16(
        /*neg_a=*/false, av, /*neg_b=*/false, bv,
        /*c_mod=*/(short)0, acc, /*reuse_a=*/false, /*reuse_b=*/false);
  }
  const float bb = bias ? bias[col0 + lm] : 0.0f;
  float* cp = C + (size_t)(row0 + hh * 8) * M + col0 + lm;
#pragma unroll
  for (int v = 0; v < 8; ++v) cp[(size_t)v * M] = acc[v] + bb;
}

// ---------------- weight transpose+convert: Wt[m*K+k] = bf16(W[k*M+m]) ------
__global__ void cvt_w_t(const float* __restrict__ W, bf16* __restrict__ Wt,
                        int K, int M) {
  int idx = blockIdx.x * 256 + threadIdx.x;
  if (idx >= K * M) return;
  int m = idx / K, k = idx - m * K;
  Wt[idx] = (bf16)W[(size_t)k * M + m];
}

// ---------------- f32 -> bf16 with optional per-col bias(es) + relu ---------
__global__ void cvt_act(const float* __restrict__ in, const float* __restrict__ b0,
                        const float* __restrict__ b1, int do_relu,
                        bf16* __restrict__ out, int n, int C) {
  int idx = blockIdx.x * 256 + threadIdx.x;
  if (idx >= n) return;
  float v = in[idx];
  int c = idx % C;
  if (b0) v += b0[c];
  if (b1) v += b1[c];
  if (do_relu) v = fmaxf(v, 0.0f);
  out[idx] = (bf16)v;
}

__global__ void fill_u32(unsigned* __restrict__ p, unsigned val, int n) {
  int idx = blockIdx.x * 256 + threadIdx.x;
  if (idx < n) p[idx] = val;
}

// order-preserving float<->uint map for atomic segment-max
__device__ __forceinline__ unsigned enc_f32(float f) {
  unsigned u = __float_as_uint(f);
  return (u & 0x80000000u) ? ~u : (u | 0x80000000u);
}
__device__ __forceinline__ float dec_f32(unsigned u) {
  unsigned b = (u & 0x80000000u) ? (u & 0x7FFFFFFFu) : ~u;
  return __uint_as_float(b);
}

// ---------------- pass A: edge scores + segment max (wave per edge) ---------
__global__ void edge_score(const int* __restrict__ src, const int* __restrict__ dst,
                           const float* __restrict__ xl, const float* __restrict__ xr,
                           const float* __restrict__ att,
                           float* __restrict__ score, unsigned* __restrict__ menc,
                           int E) {
  int e = blockIdx.x * 8 + (threadIdx.x >> 5);
  if (e >= E) return;
  int lane = threadIdx.x & 31;
  int s = src[e], d = dst[e];
  const float4* xl4 = (const float4*)(xl + (size_t)s * HC);
  const float4* xr4 = (const float4*)(xr + (size_t)d * HC);
  const float4* at4 = (const float4*)att;
#pragma unroll
  for (int h = 0; h < 2; ++h) {
    float4 a = xl4[h * 32 + lane];
    float4 b = xr4[h * 32 + lane];
    float4 t = at4[h * 32 + lane];
    float r = 0.0f, v;
    v = a.x + b.x; r += (v > 0.0f ? v : 0.2f * v) * t.x;
    v = a.y + b.y; r += (v > 0.0f ? v : 0.2f * v) * t.y;
    v = a.z + b.z; r += (v > 0.0f ? v : 0.2f * v) * t.z;
    v = a.w + b.w; r += (v > 0.0f ? v : 0.2f * v) * t.w;
#pragma unroll
    for (int o = 16; o > 0; o >>= 1) r += __shfl_xor(r, o, 32);
    if (lane == 0) {
      score[(size_t)e * 2 + h] = r;
      atomicMax(&menc[(size_t)d * 2 + h], enc_f32(r));
    }
  }
}

// ---------------- pass B: exp(score - max), segment sum ---------------------
__global__ void edge_exp_den(const int* __restrict__ dst, const float* __restrict__ score,
                             const unsigned* __restrict__ menc,
                             float* __restrict__ ex, float* __restrict__ den, int E) {
  int e = blockIdx.x * 256 + threadIdx.x;
  if (e >= E) return;
  int d = dst[e];
#pragma unroll
  for (int h = 0; h < 2; ++h) {
    unsigned u = menc[(size_t)d * 2 + h];
    float m = (u == ENC_NEG_INF) ? 0.0f : dec_f32(u);
    float v = __expf(score[(size_t)e * 2 + h] - m);
    ex[(size_t)e * 2 + h] = v;
    atomicAdd(&den[(size_t)d * 2 + h], v);
  }
}

// ---------------- pass C: acc[dst] += mean_h(alpha_h * xl[src,h]) -----------
__global__ void edge_agg(const int* __restrict__ src, const int* __restrict__ dst,
                         const float* __restrict__ xl, const float* __restrict__ ex,
                         const float* __restrict__ den, float* __restrict__ acc, int E) {
  int e = blockIdx.x * 8 + (threadIdx.x >> 5);
  if (e >= E) return;
  int lane = threadIdx.x & 31;
  int s = src[e], d = dst[e];
  float a0 = ex[(size_t)e * 2 + 0] / (den[(size_t)d * 2 + 0] + 1e-16f);
  float a1 = ex[(size_t)e * 2 + 1] / (den[(size_t)d * 2 + 1] + 1e-16f);
  const float4* xl4 = (const float4*)(xl + (size_t)s * HC);
  float4 x0 = xl4[lane];
  float4 x1 = xl4[32 + lane];
  float* ap = acc + (size_t)d * HIDC + lane * 4;
  atomicAdd(ap + 0, 0.5f * (a0 * x0.x + a1 * x1.x));
  atomicAdd(ap + 1, 0.5f * (a0 * x0.y + a1 * x1.y));
  atomicAdd(ap + 2, 0.5f * (a0 * x0.z + a1 * x1.z));
  atomicAdd(ap + 3, 0.5f * (a0 * x0.w + a1 * x1.w));
}

// ---------------- fused bias + relu + batch pool of gene nodes --------------
__global__ void pool_gene(const float* __restrict__ acc, const float* __restrict__ bias,
                          const int* __restrict__ batch, float* __restrict__ pooled,
                          int n) {
  int idx = blockIdx.x * 256 + threadIdx.x;
  if (idx >= n) return;
  int node = idx >> 7, c = idx & 127;
  float v = fmaxf(acc[idx] + bias[c], 0.0f);
  atomicAdd(&pooled[(size_t)batch[node] * HIDC + c], v);
}

extern "C" void kernel_launch(void* const* d_in, const int* in_sizes, int n_in,
                              void* d_out, int out_size, void* d_ws, size_t ws_size,
                              hipStream_t stream) {
  (void)in_sizes; (void)n_in; (void)out_size; (void)ws_size;
  const int N_SNP = 200000, N_HAP = 50000, N_GENE = 20000;
  const int E_SH = 600000, E_HG = 200000;

  const float* x_snp  = (const float*)d_in[0];
  const float* x_hap  = (const float*)d_in[1];
  const float* x_gene = (const float*)d_in[2];
  const int* e_sh_src = (const int*)d_in[3];
  const int* e_sh_dst = (const int*)d_in[4];
  const int* e_hg_src = (const int*)d_in[5];
  const int* e_hg_dst = (const int*)d_in[6];
  const int* e_gh_src = (const int*)d_in[9];
  const int* e_gh_dst = (const int*)d_in[10];
  const int* batch_gene = (const int*)d_in[11];

  // params flattened as jax pytree leaves (dict keys sorted alphabetically):
  // convs[0..1] (et: gene__hap,hap__gene,hap__snp,snp__hap; leaves Wl,Wr,att,bias,bl,br),
  // lin (gene,hap,snp; W,b), out (W,b).  Leaves start at d_in[12].
  auto P = [&](int i) { return (const float*)d_in[i]; };
  // layer 0
  const float *Wl0_gh = P(12), *Wr0_gh = P(13), *att0_gh = P(14), *bias0_gh = P(15),
              *bl0_gh = P(16), *br0_gh = P(17);
  const float *Wl0_hg = P(18), *Wr0_hg = P(19), *att0_hg = P(20), *bias0_hg = P(21),
              *bl0_hg = P(22), *br0_hg = P(23);
  const float *Wl0_sh = P(30), *Wr0_sh = P(31), *att0_sh = P(32), *bias0_sh = P(33),
              *bl0_sh = P(34), *br0_sh = P(35);
  // layer 1 (only hap__gene is live)
  const float *Wl1_hg = P(42), *Wr1_hg = P(43), *att1_hg = P(44), *bias1_hg = P(45),
              *bl1_hg = P(46), *br1_hg = P(47);
  const float *lin_gene_W = P(60), *lin_gene_b = P(61);
  const float *lin_hap_W  = P(62), *lin_hap_b  = P(63);
  const float *lin_snp_W  = P(64), *lin_snp_b  = P(65);
  const float *out_W = P(66), *out_b = P(67);

  // ---- workspace carve-up ----
  char* wsb = (char*)d_ws;
  size_t off = 0;
  auto alloc = [&](size_t bytes) -> void* {
    off = (off + 255) & ~(size_t)255;
    void* p = wsb + off; off += bytes; return p;
  };
  bf16* xb_snp  = (bf16*)alloc((size_t)N_SNP  * HIDC * 2);
  bf16* xb_hap  = (bf16*)alloc((size_t)N_HAP  * HIDC * 2);
  bf16* xb_gene = (bf16*)alloc((size_t)N_GENE * HIDC * 2);
  float* xlb = (float*)alloc((size_t)N_SNP * HC * 4);   // xl / lin staging
  float* xrb = (float*)alloc((size_t)N_HAP * HC * 4);   // xr / raw bf16 staging
  float* acc_hap  = (float*)alloc((size_t)N_HAP  * HIDC * 4);
  float* acc_gene = (float*)alloc((size_t)N_GENE * HIDC * 4);
  float* score = (float*)alloc((size_t)E_SH * 2 * 4);
  float* exb   = (float*)alloc((size_t)E_SH * 2 * 4);
  unsigned* menc = (unsigned*)alloc((size_t)N_HAP * 2 * 4);
  float* den = (float*)alloc((size_t)N_HAP * 2 * 4);
  float* pooled = (float*)alloc(32 * 128 * 4);
  bf16* pooled_b = (bf16*)alloc(32 * 128 * 2);
  bf16* wt_lin_snp  = (bf16*)alloc(64 * 128 * 2);
  bf16* wt_lin_hap  = (bf16*)alloc(64 * 128 * 2);
  bf16* wt_lin_gene = (bf16*)alloc(64 * 128 * 2);
  bf16* wt0_sh_l = (bf16*)alloc(128 * 256 * 2);
  bf16* wt0_sh_r = (bf16*)alloc(128 * 256 * 2);
  bf16* wt0_gh_l = (bf16*)alloc(128 * 256 * 2);
  bf16* wt0_gh_r = (bf16*)alloc(128 * 256 * 2);
  bf16* wt0_hg_l = (bf16*)alloc(128 * 256 * 2);
  bf16* wt0_hg_r = (bf16*)alloc(128 * 256 * 2);
  bf16* wt1_hg_l = (bf16*)alloc(128 * 256 * 2);
  bf16* wt1_hg_r = (bf16*)alloc(128 * 256 * 2);
  bf16* wt_out = (bf16*)alloc(128 * 128 * 2);

  auto cw = [&](const float* W, bf16* Wt, int K, int M) {
    int n = K * M;
    cvt_w_t<<<dim3((n + 255) / 256), dim3(256), 0, stream>>>(W, Wt, K, M);
  };
  cw(lin_snp_W, wt_lin_snp, 64, 128);
  cw(lin_hap_W, wt_lin_hap, 64, 128);
  cw(lin_gene_W, wt_lin_gene, 64, 128);
  cw(Wl0_sh, wt0_sh_l, 128, 256); cw(Wr0_sh, wt0_sh_r, 128, 256);
  cw(Wl0_gh, wt0_gh_l, 128, 256); cw(Wr0_gh, wt0_gh_r, 128, 256);
  cw(Wl0_hg, wt0_hg_l, 128, 256); cw(Wr0_hg, wt0_hg_r, 128, 256);
  cw(Wl1_hg, wt1_hg_l, 128, 256); cw(Wr1_hg, wt1_hg_r, 128, 256);
  cw(out_W, wt_out, 128, 128);

  // ---- initial per-type linears: bf16(x_raw) @ W + b -> bf16 features ----
  auto lin = [&](const float* xraw, int N, bf16* wt, const float* b, bf16* xb) {
    bf16* stage = (bf16*)xrb;
    int n0 = N * 64;
    cvt_act<<<(n0 + 255) / 256, 256, 0, stream>>>(xraw, nullptr, nullptr, 0, stage, n0, 64);
    gemm_bf16<<<dim3(N / 16, 2), dim3(32, 4), 0, stream>>>(stage, wt, b, xlb, 64, 128);
    int n1 = N * 128;
    cvt_act<<<(n1 + 255) / 256, 256, 0, stream>>>(xlb, nullptr, nullptr, 0, xb, n1, 128);
  };
  lin(x_snp, N_SNP, wt_lin_snp, lin_snp_b, xb_snp);
  lin(x_hap, N_HAP, wt_lin_hap, lin_hap_b, xb_hap);
  lin(x_gene, N_GENE, wt_lin_gene, lin_gene_b, xb_gene);

  // ---- one GATv2 conv over one edge type ----
  auto run_gat = [&](int Ns, int Nd, int E, const int* src, const int* dstp,
                     const bf16* xs, const bf16* xd,
                     const bf16* wlT, const float* bl, const bf16* wrT, const float* br,
                     const float* att, float* accd) {
    gemm_bf16<<<dim3(Ns / 16, 4), dim3(32, 4), 0, stream>>>(xs, wlT, bl, xlb, 128, 256);
    gemm_bf16<<<dim3(Nd / 16, 4), dim3(32, 4), 0, stream>>>(xd, wrT, br, xrb, 128, 256);
    fill_u32<<<(Nd * 2 + 255) / 256, 256, 0, stream>>>(menc, ENC_NEG_INF, Nd * 2);
    hipMemsetAsync(den, 0, (size_t)Nd * 2 * 4, stream);
    edge_score<<<(E + 7) / 8, 256, 0, stream>>>(src, dstp, xlb, xrb, att, score, menc, E);
    edge_exp_den<<<(E + 255) / 256, 256, 0, stream>>>(dstp, score, menc, exb, den, E);
    edge_agg<<<(E + 7) / 8, 256, 0, stream>>>(src, dstp, xlb, exb, den, accd, E);
  };

  // ---- layer 1 (live dsts: hap, gene) ----
  hipMemsetAsync(acc_hap, 0, (size_t)N_HAP * HIDC * 4, stream);
  hipMemsetAsync(acc_gene, 0, (size_t)N_GENE * HIDC * 4, stream);
  run_gat(N_SNP, N_HAP, E_SH, e_sh_src, e_sh_dst, xb_snp, xb_hap,
          wt0_sh_l, bl0_sh, wt0_sh_r, br0_sh, att0_sh, acc_hap);
  run_gat(N_GENE, N_HAP, E_HG, e_gh_src, e_gh_dst, xb_gene, xb_hap,
          wt0_gh_l, bl0_gh, wt0_gh_r, br0_gh, att0_gh, acc_hap);
  run_gat(N_HAP, N_GENE, E_HG, e_hg_src, e_hg_dst, xb_hap, xb_gene,
          wt0_hg_l, bl0_hg, wt0_hg_r, br0_hg, att0_hg, acc_gene);
  cvt_act<<<(N_HAP * 128 + 255) / 256, 256, 0, stream>>>(
      acc_hap, bias0_sh, bias0_gh, 1, xb_hap, N_HAP * 128, 128);
  cvt_act<<<(N_GENE * 128 + 255) / 256, 256, 0, stream>>>(
      acc_gene, bias0_hg, nullptr, 1, xb_gene, N_GENE * 128, 128);

  // ---- layer 2 (only hap__gene feeds the output) ----
  hipMemsetAsync(acc_gene, 0, (size_t)N_GENE * HIDC * 4, stream);
  run_gat(N_HAP, N_GENE, E_HG, e_hg_src, e_hg_dst, xb_hap, xb_gene,
          wt1_hg_l, bl1_hg, wt1_hg_r, br1_hg, att1_hg, acc_gene);

  // ---- fused bias+relu+pool, then output head ----
  hipMemsetAsync(pooled, 0, 32 * 128 * 4, stream);
  pool_gene<<<(N_GENE * 128 + 255) / 256, 256, 0, stream>>>(
      acc_gene, bias1_hg, batch_gene, pooled, N_GENE * 128);
  cvt_act<<<(4096 + 255) / 256, 256, 0, stream>>>(pooled, nullptr, nullptr, 0,
                                                  pooled_b, 4096, 128);
  gemm_bf16<<<dim3(2, 2), dim3(32, 4), 0, stream>>>(pooled_b, wt_out, out_b,
                                                    (float*)d_out, 128, 128);
}
